// mclSTExp_Attention_39350490366713
// MI455X (gfx1250) — matrix operand order repaired and checked by reading, not compiled, not run on hardware
//
#include <hip/hip_runtime.h>
#include <hip/hip_bf16.h>
#include <math.h>

typedef __attribute__((ext_vector_type(16))) _Float16 v16h;
typedef __attribute__((ext_vector_type(8)))  float    v8f;

#define NB     2048
#define SPOTD  1024
#define IMGD   1024
#define PROJD  256
#define NH     8
#define DHD    64
#define INNERD 512
#define MLPD   2048
#define QKV3   1536

__device__ __forceinline__ float gelu_exact(float x) {
  return 0.5f * x * (1.0f + erff(x * 0.70710678118654752f));
}

// ---------------------------------------------------------------------------
// CDNA5 async memory->LDS copy (GLOBAL_LOAD_ASYNC_TO_LDS_B128, ASYNCcnt).
// LDS operand is the 32-bit offset within the workgroup's LDS allocation.
// ---------------------------------------------------------------------------
__device__ __forceinline__ void async_copy_b128(void* lds_dst, const void* gsrc) {
  unsigned loff = (unsigned)(unsigned long long)
      (__attribute__((address_space(3))) char*)lds_dst;
  asm volatile("global_load_async_to_lds_b128 %0, %1, off"
               :: "v"(loff), "v"(gsrc) : "memory");
}

__device__ __forceinline__ void wait_asynccnt0() {
#if __has_builtin(__builtin_amdgcn_s_wait_asynccnt)
  __builtin_amdgcn_s_wait_asynccnt(0);
#else
  asm volatile("s_wait_asynccnt 0x0" ::: "memory");
#endif
}

// ---------------------------------------------------------------------------
// spot = expression + x_table[x_idx] + y_table[y_idx]
// ---------------------------------------------------------------------------
__global__ __launch_bounds__(256)
void embed_kernel(const float* __restrict__ expr, const float* __restrict__ xt,
                  const float* __restrict__ yt, const int* __restrict__ xi,
                  const int* __restrict__ yi, float* __restrict__ spot) {
  int b = blockIdx.x;
  long xb = xi[b], yb = yi[b];
  for (int s = threadIdx.x; s < SPOTD; s += blockDim.x)
    spot[(long)b * SPOTD + s] =
        expr[(long)b * SPOTD + s] + xt[xb * SPOTD + s] + yt[yb * SPOTD + s];
}

// ---------------------------------------------------------------------------
// row-wise layernorm; optional residual input (out = LN(x + res))
// ---------------------------------------------------------------------------
__global__ __launch_bounds__(256)
void layernorm_kernel(const float* __restrict__ x, const float* __restrict__ res,
                      const float* __restrict__ g, const float* __restrict__ bta,
                      float* __restrict__ out, int D) {
  int row = blockIdx.x;
  __shared__ float red[2][8];
  const float* xr = x + (long)row * D;
  const float* rr = res ? res + (long)row * D : nullptr;
  float s = 0.f, s2 = 0.f;
  for (int i = threadIdx.x; i < D; i += blockDim.x) {
    float v = xr[i] + (rr ? rr[i] : 0.f);
    s += v; s2 += v * v;
  }
  for (int off = 16; off; off >>= 1) { s += __shfl_xor(s, off); s2 += __shfl_xor(s2, off); }
  int wid = threadIdx.x >> 5;
  if ((threadIdx.x & 31) == 0) { red[0][wid] = s; red[1][wid] = s2; }
  __syncthreads();
  if (threadIdx.x == 0) {
    float a = 0.f, b2 = 0.f;
    for (int i = 0; i < 8; ++i) { a += red[0][i]; b2 += red[1][i]; }
    red[0][0] = a; red[1][0] = b2;
  }
  __syncthreads();
  float mean = red[0][0] / (float)D;
  float var  = red[1][0] / (float)D - mean * mean;
  float rstd = rsqrtf(var + 1e-5f);
  for (int i = threadIdx.x; i < D; i += blockDim.x) {
    float v = xr[i] + (rr ? rr[i] : 0.f);
    out[(long)row * D + i] = (v - mean) * rstd * g[i] + bta[i];
  }
}

// ---------------------------------------------------------------------------
// WMMA GEMM: Out[M,N] = act( A[M,K] @ B + bias + Res )
//   TRANSB: B is [N,K] (compute A @ B^T), else B is [K,N]
//   GELU_A: apply exact gelu to A elements at fragment build
//   ACT==1: exact gelu epilogue
// Block tile 128x64, BK=32, 8 waves, each wave a 32x32 tile (4 WMMA/k-step).
// f32 tiles staged to LDS via global_load_async_to_lds_b128; converted to f16
// while building fragments. Dims must be multiples of 128 (M), 64 (N), 32 (K).
// ---------------------------------------------------------------------------
template<int TRANSB, int GELU_A, int ACT>
__global__ __launch_bounds__(256)
void gemm_wmma(const float* __restrict__ A, const float* __restrict__ Bm,
               const float* __restrict__ bias, const float* __restrict__ Res,
               float* __restrict__ Out, int M, int N, int K) {
  __shared__ float Asf[128 * 32];                // Asf[m][k]
  __shared__ float Bsf[64 * 32];                 // TRANSB: Bsf[n][k], else Bsf[k][n]
  int m0 = blockIdx.y * 128;
  int n0 = blockIdx.x * 64;
  int tid  = threadIdx.x;
  int wave = tid >> 5;
  int lane = tid & 31;
  int wm = wave >> 1, wn = wave & 1;
  int hi = lane >> 4, lo = lane & 15;
  v8f c00 = {}, c01 = {}, c10 = {}, c11 = {};
  for (int k0 = 0; k0 < K; k0 += 32) {
    // ---- async-stage A tile (128x32 f32): 1024 16B chunks / 256 threads ----
#pragma unroll
    for (int i = 0; i < 4; ++i) {
      int c = tid + i * 256;
      int row = c >> 3, col = (c & 7) * 4;
      async_copy_b128(&Asf[row * 32 + col], A + (long)(m0 + row) * K + k0 + col);
    }
    // ---- async-stage B tile (64x32 f32): 512 chunks / 256 threads ----
#pragma unroll
    for (int i = 0; i < 2; ++i) {
      int c = tid + i * 256;
      if (TRANSB) {
        int n = c >> 3, kk = (c & 7) * 4;
        async_copy_b128(&Bsf[n * 32 + kk], Bm + (long)(n0 + n) * K + k0 + kk);
      } else {
        int kk = c >> 4, n = (c & 15) * 4;
        async_copy_b128(&Bsf[kk * 64 + n], Bm + (long)(k0 + kk) * N + n0 + n);
      }
    }
    // prefetch next K tile into cache while async copies land
    if (k0 + 32 < K) {
      __builtin_prefetch(A + (long)(m0 + (tid >> 3)) * K + (k0 + 32) + (tid & 7) * 4, 0, 1);
      __builtin_prefetch(TRANSB ? Bm + (long)(n0 + (tid >> 3)) * K + (k0 + 32)
                                : Bm + (long)(k0 + 32 + (tid >> 4)) * N + n0 + (tid & 15) * 4,
                         0, 1);
    }
    wait_asynccnt0();
    __syncthreads();
    // ---- fragments (f32 LDS -> f16 regs) ----
    v16h a0, a1, b0, b1;
#pragma unroll
    for (int e = 0; e < 16; ++e) {
      int kk = hi * 8 + (e < 8 ? e : e + 8);     // 16-bit A layout (16x32)
      float va0 = Asf[(wm * 32 + lo) * 32 + kk];
      float va1 = Asf[(wm * 32 + 16 + lo) * 32 + kk];
      if (GELU_A) { va0 = gelu_exact(va0); va1 = gelu_exact(va1); }
      a0[e] = (_Float16)va0;
      a1[e] = (_Float16)va1;
    }
#pragma unroll
    for (int e = 0; e < 16; ++e) {
      int kk = hi * 8 + (e < 8 ? e : e + 8);      // 16-bit B layout (32x16)
      int nA = wn * 32 + lo, nBt = nA + 16;
      b0[e] = (_Float16)(TRANSB ? Bsf[nA * 32 + kk]  : Bsf[kk * 64 + nA]);
      b1[e] = (_Float16)(TRANSB ? Bsf[nBt * 32 + kk] : Bsf[kk * 64 + nBt]);
    }
    c00 = __builtin_amdgcn_wmma_f32_16x16x32_f16(false, a0, false, b0, (short)0, c00, false, false);
    c01 = __builtin_amdgcn_wmma_f32_16x16x32_f16(false, a0, false, b1, (short)0, c01, false, false);
    c10 = __builtin_amdgcn_wmma_f32_16x16x32_f16(false, a1, false, b0, (short)0, c10, false, false);
    c11 = __builtin_amdgcn_wmma_f32_16x16x32_f16(false, a1, false, b1, (short)0, c11, false, false);
    __syncthreads();
  }
  // ---- epilogue (C layout: M = 8*hi + r, N = lo) ----
#pragma unroll
  for (int r = 0; r < 8; ++r) {
    int row0 = m0 + wm * 32 + hi * 8 + r;
    int row1 = row0 + 16;
    int nA = n0 + wn * 32 + lo;
    int nBt = nA + 16;
    float v00 = c00[r], v01 = c01[r], v10 = c10[r], v11 = c11[r];
    if (bias) {
      float bA = bias[nA], bB = bias[nBt];
      v00 += bA; v01 += bB; v10 += bA; v11 += bB;
    }
    if (Res) {
      v00 += Res[(long)row0 * N + nA];  v01 += Res[(long)row0 * N + nBt];
      v10 += Res[(long)row1 * N + nA];  v11 += Res[(long)row1 * N + nBt];
    }
    if (ACT == 1) {
      v00 = gelu_exact(v00); v01 = gelu_exact(v01);
      v10 = gelu_exact(v10); v11 = gelu_exact(v11);
    }
    Out[(long)row0 * N + nA]  = v00;
    Out[(long)row0 * N + nBt] = v01;
    Out[(long)row1 * N + nA]  = v10;
    Out[(long)row1 * N + nBt] = v11;
  }
}

// ---------------------------------------------------------------------------
// Flash attention over qkv[B, 1536] (q|k|v, head dim 64, 8 heads, scale 1/8).
// One wave per (head, 16-query tile); 4 waves per block. Online softmax.
// Writes o_out[B, 512] (= o.transpose(1,0,2).reshape).
// ---------------------------------------------------------------------------
__global__ __launch_bounds__(128)
void attn_kernel(const float* __restrict__ qkv, float* __restrict__ o_out) {
  __shared__ _Float16 P[4][16 * 32];
  int wave = threadIdx.x >> 5;
  int lane = threadIdx.x & 31;
  int hi = lane >> 4, lo = lane & 15;
  int h  = blockIdx.y;
  int q0 = (blockIdx.x * 4 + wave) * 16;
  const float scale = 0.125f;  // DH^-0.5

  // Q fragments (scale folded in): d 0..31 and 32..63
  v16h qa0, qa1;
  {
    const float* qp = qkv + (long)(q0 + lo) * QKV3 + h * DHD;
#pragma unroll
    for (int e = 0; e < 16; ++e) {
      int d = hi * 16 + e;
      qa0[e] = (_Float16)(qp[d] * scale);
      qa1[e] = (_Float16)(qp[d + 32] * scale);
    }
  }

  float mrow[8], lrow[8];
  v8f o0 = {}, o1 = {}, o2 = {}, o3 = {};
#pragma unroll
  for (int r = 0; r < 8; ++r) { mrow[r] = -3.0e38f; lrow[r] = 0.f; }

  _Float16* Pw = P[wave];
  for (int j0 = 0; j0 < NB; j0 += 32) {
    // ---- S = Q @ K^T for 32 keys (two 16x16 tiles) ----
    v8f s0 = {}, s1 = {};
    {
      v16h kb00, kb01, kb10, kb11;
      const float* kp0 = qkv + (long)(j0 + lo) * QKV3 + INNERD + h * DHD + hi * 16;
      const float* kp1 = qkv + (long)(j0 + 16 + lo) * QKV3 + INNERD + h * DHD + hi * 16;
#pragma unroll
      for (int e = 0; e < 16; ++e) {
        kb00[e] = (_Float16)kp0[e];      kb01[e] = (_Float16)kp0[e + 32];
        kb10[e] = (_Float16)kp1[e];      kb11[e] = (_Float16)kp1[e + 32];
      }
      s0 = __builtin_amdgcn_wmma_f32_16x16x32_f16(false, qa0, false, kb00, (short)0, s0, false, false);
      s0 = __builtin_amdgcn_wmma_f32_16x16x32_f16(false, qa1, false, kb01, (short)0, s0, false, false);
      s1 = __builtin_amdgcn_wmma_f32_16x16x32_f16(false, qa0, false, kb10, (short)0, s1, false, false);
      s1 = __builtin_amdgcn_wmma_f32_16x16x32_f16(false, qa1, false, kb11, (short)0, s1, false, false);
    }
    // ---- online softmax update (rows M = 8*hi + r, cols across 16 lanes) ----
    float p0[8], p1[8];
#pragma unroll
    for (int r = 0; r < 8; ++r) {
      float c = fmaxf(s0[r], s1[r]);
      for (int off = 1; off < 16; off <<= 1) c = fmaxf(c, __shfl_xor(c, off));
      float nm = fmaxf(mrow[r], c);
      float f  = expf(mrow[r] - nm);
      p0[r] = expf(s0[r] - nm);
      p1[r] = expf(s1[r] - nm);
      float rsum = p0[r] + p1[r];
      for (int off = 1; off < 16; off <<= 1) rsum += __shfl_xor(rsum, off);
      lrow[r] = lrow[r] * f + rsum;
      mrow[r] = nm;
      o0[r] *= f; o1[r] *= f; o2[r] *= f; o3[r] *= f;
    }
    // ---- stage P (16x32) to LDS, reload as A fragment ----
#pragma unroll
    for (int r = 0; r < 8; ++r) {
      Pw[(hi * 8 + r) * 32 + lo]      = (_Float16)p0[r];
      Pw[(hi * 8 + r) * 32 + 16 + lo] = (_Float16)p1[r];
    }
    __syncthreads();
    v16h pa;
#pragma unroll
    for (int e = 0; e < 16; ++e) {
      int kk = hi * 16 + e;
      pa[e] = Pw[lo * 32 + kk];
    }
    // ---- O += P @ V (four 16x16 d-tiles) ----
    const float* vp = qkv + (long)(j0 + hi * 16) * QKV3 + 2 * INNERD + h * DHD + lo;
    v16h vb;
#pragma unroll
    for (int e = 0; e < 16; ++e) vb[e] = (_Float16)vp[(long)e * QKV3];
    o0 = __builtin_amdgcn_wmma_f32_16x16x32_f16(false, pa, false, vb, (short)0, o0, false, false);
#pragma unroll
    for (int e = 0; e < 16; ++e) vb[e] = (_Float16)vp[(long)e * QKV3 + 16];
    o1 = __builtin_amdgcn_wmma_f32_16x16x32_f16(false, pa, false, vb, (short)0, o1, false, false);
#pragma unroll
    for (int e = 0; e < 16; ++e) vb[e] = (_Float16)vp[(long)e * QKV3 + 32];
    o2 = __builtin_amdgcn_wmma_f32_16x16x32_f16(false, pa, false, vb, (short)0, o2, false, false);
#pragma unroll
    for (int e = 0; e < 16; ++e) vb[e] = (_Float16)vp[(long)e * QKV3 + 48];
    o3 = __builtin_amdgcn_wmma_f32_16x16x32_f16(false, pa, false, vb, (short)0, o3, false, false);
    __syncthreads();
  }
  // ---- write O / l ----
  float* op = o_out + (long)(q0 + hi * 8) * INNERD + h * DHD + lo;
#pragma unroll
  for (int r = 0; r < 8; ++r) {
    float inv = 1.f / lrow[r];
    op[(long)r * INNERD +  0] = o0[r] * inv;
    op[(long)r * INNERD + 16] = o1[r] * inv;
    op[(long)r * INNERD + 32] = o2[r] * inv;
    op[(long)r * INNERD + 48] = o3[r] * inv;
  }
}

// ---------------------------------------------------------------------------
// per-row (or per-column) max + sum(exp(x-max)) of scale*X  (for softmax/LSE)
// ---------------------------------------------------------------------------
__global__ __launch_bounds__(256)
void stats_kernel(const float* __restrict__ X, float scale, int colmode,
                  float* __restrict__ mout, float* __restrict__ sout, int N) {
  int i = blockIdx.x;
  __shared__ float sh[8];
  float m = -3.0e38f;
  for (int j = threadIdx.x; j < N; j += blockDim.x) {
    float v = scale * (colmode ? X[(long)j * N + i] : X[(long)i * N + j]);
    m = fmaxf(m, v);
  }
  for (int off = 16; off; off >>= 1) m = fmaxf(m, __shfl_xor(m, off));
  if ((threadIdx.x & 31) == 0) sh[threadIdx.x >> 5] = m;
  __syncthreads();
  if (threadIdx.x < 8) {
    float t = sh[threadIdx.x];
    for (int off = 4; off; off >>= 1) t = fmaxf(t, __shfl_xor(t, off));
    if (threadIdx.x == 0) sh[0] = t;
  }
  __syncthreads();
  m = sh[0];
  __syncthreads();
  float s = 0.f;
  for (int j = threadIdx.x; j < N; j += blockDim.x) {
    float v = scale * (colmode ? X[(long)j * N + i] : X[(long)i * N + j]);
    s += expf(v - m);
  }
  for (int off = 16; off; off >>= 1) s += __shfl_xor(s, off);
  if ((threadIdx.x & 31) == 0) sh[threadIdx.x >> 5] = s;
  __syncthreads();
  if (threadIdx.x == 0) {
    float t = 0.f;
    for (int w = 0; w < 8; ++w) t += sh[w];
    mout[i] = m; sout[i] = t;
  }
}

// ---------------------------------------------------------------------------
// partial[i] = sum_j targets[i][j] * (2*logits[i][j] - rowlse[i] - collse[j])
// ---------------------------------------------------------------------------
__global__ __launch_bounds__(256)
void loss_partial_kernel(const float* __restrict__ logits, const float* __restrict__ G,
                         const float* __restrict__ rm, const float* __restrict__ rs,
                         const float* __restrict__ cm, const float* __restrict__ cs,
                         const float* __restrict__ tm, const float* __restrict__ ts,
                         float* __restrict__ partial, int N) {
  int i = blockIdx.x;
  __shared__ float sh[8];
  float rowlse = rm[i] + logf(rs[i]);
  float tmi = tm[i];
  float inv_tsi = 1.f / ts[i];
  float acc = 0.f;
  for (int j = threadIdx.x; j < N; j += blockDim.x) {
    float lg = logits[(long)i * N + j];
    float collse = cm[j] + logf(cs[j]);
    float t = expf(0.5f * G[(long)i * N + j] - tmi) * inv_tsi;
    acc += t * (2.f * lg - rowlse - collse);
  }
  for (int off = 16; off; off >>= 1) acc += __shfl_xor(acc, off);
  if ((threadIdx.x & 31) == 0) sh[threadIdx.x >> 5] = acc;
  __syncthreads();
  if (threadIdx.x == 0) {
    float t = 0.f;
    for (int w = 0; w < 8; ++w) t += sh[w];
    partial[i] = t;
  }
}

__global__ __launch_bounds__(256)
void loss_final_kernel(const float* __restrict__ partial, float* __restrict__ out, int N) {
  __shared__ float sh[8];
  float acc = 0.f;
  for (int i = threadIdx.x; i < N; i += blockDim.x) acc += partial[i];
  for (int off = 16; off; off >>= 1) acc += __shfl_xor(acc, off);
  if ((threadIdx.x & 31) == 0) sh[threadIdx.x >> 5] = acc;
  __syncthreads();
  if (threadIdx.x == 0) {
    float t = 0.f;
    for (int w = 0; w < 8; ++w) t += sh[w];
    out[0] = -(0.5f / (float)N) * t;
  }
}

// ---------------------------------------------------------------------------
extern "C" void kernel_launch(void* const* d_in, const int* in_sizes, int n_in,
                              void* d_out, int out_size, void* d_ws, size_t ws_size,
                              hipStream_t stream) {
  (void)in_sizes; (void)n_in; (void)out_size; (void)ws_size;
  const float* image_features = (const float*)d_in[0];
  const float* expression     = (const float*)d_in[1];
  const float* x_table        = (const float*)d_in[2];
  const float* y_table        = (const float*)d_in[3];
  const float* ln1_g = (const float*)d_in[4];
  const float* ln1_b = (const float*)d_in[5];
  const float* w_qkv = (const float*)d_in[6];
  const float* w_o   = (const float*)d_in[7];
  const float* b_o   = (const float*)d_in[8];
  const float* ln2_g = (const float*)d_in[9];
  const float* ln2_b = (const float*)d_in[10];
  const float* w1    = (const float*)d_in[11];
  const float* b1    = (const float*)d_in[12];
  const float* w2    = (const float*)d_in[13];
  const float* b2    = (const float*)d_in[14];
  const float* img_proj_w  = (const float*)d_in[15];
  const float* img_proj_b  = (const float*)d_in[16];
  const float* img_fc_w    = (const float*)d_in[17];
  const float* img_fc_b    = (const float*)d_in[18];
  const float* img_ln_g    = (const float*)d_in[19];
  const float* img_ln_b    = (const float*)d_in[20];
  const float* spot_proj_w = (const float*)d_in[21];
  const float* spot_proj_b = (const float*)d_in[22];
  const float* spot_fc_w   = (const float*)d_in[23];
  const float* spot_fc_b   = (const float*)d_in[24];
  const float* spot_ln_g   = (const float*)d_in[25];
  const float* spot_ln_b   = (const float*)d_in[26];
  const int*   x_idx = (const int*)d_in[27];
  const int*   y_idx = (const int*)d_in[28];

  float* W = (float*)d_ws;
  const long M1 = 1024L * 1024L;
  float* spot   = W;            // 2M floats
  float* hbuf   = W + 2 * M1;   // 2M
  float* qkvbuf = W + 4 * M1;   // 3M
  float* obuf   = W + 7 * M1;   // 1M
  float* mlpbuf = W + 8 * M1;   // 4M
  // reused after the transformer layers:
  float* img_p  = qkvbuf;
  float* img_h  = qkvbuf +  512 * 1024;
  float* img_e  = qkvbuf + 1024 * 1024;
  float* spot_p = qkvbuf + 1536 * 1024;
  float* spot_h = qkvbuf + 2048 * 1024;
  float* spot_e = qkvbuf + 2560 * 1024;
  float* logits = mlpbuf;       // 4M
  float* Gsim   = W + 12 * M1;  // 4M
  float* rm = W + 16 * M1;
  float* rs = rm + NB;  float* cm = rs + NB;  float* cs = cm + NB;
  float* tm = cs + NB;  float* ts = tm + NB;  float* partial = ts + NB;

  embed_kernel<<<NB, 256, 0, stream>>>(expression, x_table, y_table, x_idx, y_idx, spot);

  for (int l = 0; l < 2; ++l) {
    layernorm_kernel<<<NB, 256, 0, stream>>>(spot, nullptr, ln1_g + l * SPOTD, ln1_b + l * SPOTD, hbuf, SPOTD);
    gemm_wmma<0,0,0><<<dim3(QKV3 / 64, NB / 128), 256, 0, stream>>>(
        hbuf, w_qkv + (long)l * SPOTD * QKV3, nullptr, nullptr, qkvbuf, NB, QKV3, SPOTD);
    attn_kernel<<<dim3(NB / 64, NH), 128, 0, stream>>>(qkvbuf, obuf);
    gemm_wmma<0,0,0><<<dim3(SPOTD / 64, NB / 128), 256, 0, stream>>>(
        obuf, w_o + (long)l * INNERD * SPOTD, b_o + l * SPOTD, spot, spot, NB, SPOTD, INNERD);
    layernorm_kernel<<<NB, 256, 0, stream>>>(spot, nullptr, ln2_g + l * SPOTD, ln2_b + l * SPOTD, hbuf, SPOTD);
    gemm_wmma<0,0,1><<<dim3(MLPD / 64, NB / 128), 256, 0, stream>>>(
        hbuf, w1 + (long)l * SPOTD * MLPD, b1 + l * MLPD, nullptr, mlpbuf, NB, MLPD, SPOTD);
    gemm_wmma<0,0,0><<<dim3(SPOTD / 64, NB / 128), 256, 0, stream>>>(
        mlpbuf, w2 + (long)l * MLPD * SPOTD, b2 + l * SPOTD, spot, spot, NB, SPOTD, MLPD);
  }

  // projection heads
  gemm_wmma<0,0,0><<<dim3(PROJD / 64, NB / 128), 256, 0, stream>>>(
      image_features, img_proj_w, img_proj_b, nullptr, img_p, NB, PROJD, IMGD);
  gemm_wmma<0,1,0><<<dim3(PROJD / 64, NB / 128), 256, 0, stream>>>(
      img_p, img_fc_w, img_fc_b, nullptr, img_h, NB, PROJD, PROJD);
  layernorm_kernel<<<NB, 256, 0, stream>>>(img_h, img_p, img_ln_g, img_ln_b, img_e, PROJD);
  gemm_wmma<0,0,0><<<dim3(PROJD / 64, NB / 128), 256, 0, stream>>>(
      spot, spot_proj_w, spot_proj_b, nullptr, spot_p, NB, PROJD, SPOTD);
  gemm_wmma<0,1,0><<<dim3(PROJD / 64, NB / 128), 256, 0, stream>>>(
      spot_p, spot_fc_w, spot_fc_b, nullptr, spot_h, NB, PROJD, PROJD);
  layernorm_kernel<<<NB, 256, 0, stream>>>(spot_h, spot_p, spot_ln_g, spot_ln_b, spot_e, PROJD);

  // contrastive matrices
  gemm_wmma<1,0,0><<<dim3(NB / 64, NB / 128), 256, 0, stream>>>(
      spot_e, img_e, nullptr, nullptr, logits, NB, NB, PROJD);
  gemm_wmma<1,0,0><<<dim3(NB / 64, NB / 128), 256, 0, stream>>>(
      img_e, img_e, nullptr, nullptr, Gsim, NB, NB, PROJD);
  gemm_wmma<1,0,0><<<dim3(NB / 64, NB / 128), 256, 0, stream>>>(
      spot_e, spot_e, nullptr, Gsim, Gsim, NB, NB, PROJD);

  // loss
  stats_kernel<<<NB, 256, 0, stream>>>(logits, 1.0f, 0, rm, rs, NB);
  stats_kernel<<<NB, 256, 0, stream>>>(logits, 1.0f, 1, cm, cs, NB);
  stats_kernel<<<NB, 256, 0, stream>>>(Gsim, 0.5f, 0, tm, ts, NB);
  loss_partial_kernel<<<NB, 256, 0, stream>>>(logits, Gsim, rm, rs, cm, cs, tm, ts, partial, NB);
  loss_final_kernel<<<1, 256, 0, stream>>>(partial, (float*)d_out, NB);
}
